// DecoderLayer_28656021799091
// MI455X (gfx1250) — matrix operand order repaired and checked by reading, not compiled, not run on hardware
//
#include <hip/hip_runtime.h>
#include <math.h>

// Problem constants (from reference): B=2, S=2048, D=1024, H=16, HD=64, F=4096
#define S_LEN 2048
#define D_DIM 1024
#define H_NUM 16
#define HD_DIM 64
#define F_DIM 4096
#define B_NUM 2
#define M_ROWS (B_NUM * S_LEN) // 4096

typedef __attribute__((ext_vector_type(16))) __bf16 v16bf;
typedef __attribute__((ext_vector_type(8)))  float  v8f;
typedef __attribute__((ext_vector_type(4)))  unsigned int u32x4;
typedef __attribute__((ext_vector_type(4)))  float f32x4;

union FragBF { v16bf v; unsigned int u[8]; };

__device__ __forceinline__ v8f wmma_bf16(v16bf a, v16bf b, v8f c) {
  // D = A(16x32 bf16) * B(32x16 bf16) + C(16x16 f32)
  return __builtin_amdgcn_wmma_f32_16x16x32_bf16(false, a, false, b, (short)0, c, false, false);
}

// CDNA5 async global->LDS copy (16B per lane), tracked with ASYNCcnt.
// Generic LDS pointers keep the LDS byte offset in addr[31:0] (ISA 10.2),
// so truncation yields the VDST LDS-address operand.
__device__ __forceinline__ void async_copy_b128(const __bf16* g, __bf16* l) {
  unsigned lds = (unsigned)(size_t)l;
  asm volatile("global_load_async_to_lds_b128 %0, %1, off"
               :
               : "v"(lds), "v"(g)
               : "memory");
}

// ---------------------------------------------------------------------------
// Weight convert + transpose: Wt[n][k] = (bf16)W[k][n]
// ---------------------------------------------------------------------------
__global__ __launch_bounds__(256)
void cvt_transpose(const float* __restrict__ w, __bf16* __restrict__ wt, int K, int N) {
  int idx = blockIdx.x * 256 + threadIdx.x;
  int k = idx % K;
  int n = idx / K;
  wt[idx] = (__bf16)w[(size_t)k * N + n];
}

// ---------------------------------------------------------------------------
// V transpose for attention: vt[b][h][hd][s] = v[b][s][h][hd]
// ---------------------------------------------------------------------------
__global__ __launch_bounds__(256)
void vtrans(const __bf16* __restrict__ v, __bf16* __restrict__ vt) {
  int idx = blockIdx.x * 256 + threadIdx.x;
  int s  = idx & (S_LEN - 1);
  int r  = idx >> 11;              // S_LEN = 2048
  int hd = r & (HD_DIM - 1);
  r >>= 6;
  int h  = r & (H_NUM - 1);
  int b  = r >> 4;
  vt[idx] = v[((size_t)(b * S_LEN + s) * H_NUM + h) * HD_DIM + hd];
}

// ---------------------------------------------------------------------------
// LayerNorm (f32 in -> bf16 out), one 1024-wide row per 256-thread block
// ---------------------------------------------------------------------------
__global__ __launch_bounds__(256)
void layernorm_bf16(const float* __restrict__ x, const float* __restrict__ g,
                    const float* __restrict__ bt, __bf16* __restrict__ out) {
  const int row = blockIdx.x;
  const int tid = threadIdx.x;
  const float* xr = x + (size_t)row * D_DIM;
  f32x4 xv = *(const f32x4*)(xr + tid * 4);
  float s = xv.x + xv.y + xv.z + xv.w;
  float q = xv.x * xv.x + xv.y * xv.y + xv.z * xv.z + xv.w * xv.w;
#pragma unroll
  for (int m = 16; m >= 1; m >>= 1) {
    s += __shfl_xor(s, m, 32);
    q += __shfl_xor(q, m, 32);
  }
  __shared__ float wsum[8], wsq[8];
  const int lane = tid & 31, wid = tid >> 5;
  if (lane == 0) { wsum[wid] = s; wsq[wid] = q; }
  __syncthreads();
  float ts = 0.f, tq = 0.f;
#pragma unroll
  for (int i = 0; i < 8; ++i) { ts += wsum[i]; tq += wsq[i]; }
  float mu   = ts * (1.0f / D_DIM);
  float var  = tq * (1.0f / D_DIM) - mu * mu;
  float rstd = rsqrtf(var + 1e-5f);
  __bf16* orow = out + (size_t)row * D_DIM;
#pragma unroll
  for (int i = 0; i < 4; ++i) {
    int col = tid * 4 + i;
    orow[col] = (__bf16)((xv[i] - mu) * rstd * g[col] + bt[col]);
  }
}

// ---------------------------------------------------------------------------
// GEMM: C[M][N] = A[M][K](bf16) * Bt[N][K](bf16)  + fused epilogue
//  epi 0: -> bf16
//  epi 1: +bias -> bf16
//  epi 2: gelu(.+bias) -> bf16
//  epi 3: +bias +res -> f32
// Block 256 thr (8 waves), tile 128x128x32; wave computes 32x64 (2x4 tiles).
// Tiles staged with double-buffered GLOBAL_LOAD_ASYNC_TO_LDS_B128.
// ---------------------------------------------------------------------------
__global__ __launch_bounds__(256)
void gemm_bf16_wmma(const __bf16* __restrict__ A, const __bf16* __restrict__ Bt,
                    const float* __restrict__ bias, const float* __restrict__ res,
                    float* __restrict__ outF, __bf16* __restrict__ outB,
                    int N, int K, int epi) {
  __shared__ __bf16 As[2][128][40];
  __shared__ __bf16 Bs[2][128][40];
  const int tid  = threadIdx.x;
  const int lane = tid & 31, wid = tid >> 5;
  const int wm = wid & 3, wn = wid >> 2;
  const int m0 = blockIdx.y * 128, n0 = blockIdx.x * 128;
  const int lrow = tid >> 1;
  const int lcol = (tid & 1) << 4;
  const __bf16* aSrc = A  + (size_t)(m0 + lrow) * K + lcol;
  const __bf16* bSrc = Bt + (size_t)(n0 + lrow) * K + lcol;

  v8f acc[2][4] = {};
  const int mlane = lane & 15;
  const int kbA = (lane < 16) ? 0 : 8;   // A frag: K in {kbA..kbA+7, kbA+16..kbA+23}
  const int kbB = (lane < 16) ? 0 : 8;   // B frag: K in {kbB..kbB+7, kbB+16..kbB+23}
  const int T = K >> 5;

  // Stage tile 0 into buffer 0 (4 async ops per thread per tile).
  async_copy_b128(aSrc,     &As[0][lrow][lcol]);
  async_copy_b128(aSrc + 8, &As[0][lrow][lcol + 8]);
  async_copy_b128(bSrc,     &Bs[0][lrow][lcol]);
  async_copy_b128(bSrc + 8, &Bs[0][lrow][lcol + 8]);

  for (int t = 0; t < T; ++t) {
    const int cur = t & 1;
    if (t + 1 < T) {
      const int nxt = cur ^ 1;
      const __bf16* an = aSrc + (size_t)(t + 1) * 32;
      const __bf16* bn = bSrc + (size_t)(t + 1) * 32;
      async_copy_b128(an,     &As[nxt][lrow][lcol]);
      async_copy_b128(an + 8, &As[nxt][lrow][lcol + 8]);
      async_copy_b128(bn,     &Bs[nxt][lrow][lcol]);
      async_copy_b128(bn + 8, &Bs[nxt][lrow][lcol + 8]);
      // In-order completion: <=4 outstanding means tile t has landed.
      asm volatile("s_wait_asynccnt 0x4" ::: "memory");
    } else {
      asm volatile("s_wait_asynccnt 0x0" ::: "memory");
    }
    __syncthreads();  // all waves' tile t resident in LDS

    FragBF af[2], bfr[4];
#pragma unroll
    for (int i = 0; i < 2; ++i) {
      int r = wm * 32 + i * 16 + mlane;
      *(u32x4*)&af[i].u[0] = *(const u32x4*)&As[cur][r][kbA];
      *(u32x4*)&af[i].u[4] = *(const u32x4*)&As[cur][r][kbA + 16];
    }
#pragma unroll
    for (int j = 0; j < 4; ++j) {
      int r = wn * 64 + j * 16 + mlane;
      *(u32x4*)&bfr[j].u[0] = *(const u32x4*)&Bs[cur][r][kbB];
      *(u32x4*)&bfr[j].u[4] = *(const u32x4*)&Bs[cur][r][kbB + 16];
    }
#pragma unroll
    for (int i = 0; i < 2; ++i)
#pragma unroll
      for (int j = 0; j < 4; ++j)
        acc[i][j] = wmma_bf16(af[i].v, bfr[j].v, acc[i][j]);

    __syncthreads();  // buffer `cur` free for async overwrite at t+2
  }

  // Epilogue. C layout: VGPR e holds row m = e + 8*(lane>=16), col = lane&15.
  const int rbase = m0 + wm * 32 + ((lane < 16) ? 0 : 8);
#pragma unroll
  for (int i = 0; i < 2; ++i) {
#pragma unroll
    for (int j = 0; j < 4; ++j) {
      int col = n0 + wn * 64 + j * 16 + mlane;
      float bv = (epi > 0) ? bias[col] : 0.f;
#pragma unroll
      for (int e = 0; e < 8; ++e) {
        int row = rbase + i * 16 + e;
        size_t idx = (size_t)row * N + col;
        float vAcc = acc[i][j][e];
        if (epi == 0) {
          outB[idx] = (__bf16)vAcc;
        } else if (epi == 1) {
          outB[idx] = (__bf16)(vAcc + bv);
        } else if (epi == 2) {
          float t = vAcc + bv;
          outB[idx] = (__bf16)(0.5f * t * (1.0f + erff(t * 0.70710678f)));
        } else {
          outF[idx] = vAcc + bv + res[idx];
        }
      }
    }
  }
}

// ---------------------------------------------------------------------------
// Flash attention, faithful to reference mask bug: key <= query -> exactly -1e9.
// One wave per 16-row q tile of one (b,h). Q/K frags loaded directly from
// global (hd contiguous). V pre-transposed so PV B-frags are contiguous.
// ---------------------------------------------------------------------------
__global__ __launch_bounds__(128)
void attn_wmma(const __bf16* __restrict__ Q, const __bf16* __restrict__ Kc,
               const __bf16* __restrict__ Vt, __bf16* __restrict__ O) {
  __shared__ __bf16 Ps[4][16][40];
  const int tid  = threadIdx.x;
  const int lane = tid & 31, wid = tid >> 5;
  const int mlane = lane & 15, hi = lane >> 4;
  const int qt = blockIdx.x * 4 + wid;
  const int h  = blockIdx.y, b = blockIdx.z;
  const int kbA = hi ? 8 : 0;
  const int kbB = hi ? 8 : 0;

  // Q fragments: two hd-chunks of 32
  const int qrow = qt * 16 + mlane;
  const __bf16* qp = Q + ((size_t)(b * S_LEN + qrow) * H_NUM + h) * HD_DIM;
  FragBF qf[2];
#pragma unroll
  for (int c = 0; c < 2; ++c) {
    *(u32x4*)&qf[c].u[0] = *(const u32x4*)(qp + 32 * c + kbA);
    *(u32x4*)&qf[c].u[4] = *(const u32x4*)(qp + 32 * c + kbA + 16);
  }

  float rm[8], rl[8];
  v8f acc[4] = {};
#pragma unroll
  for (int e = 0; e < 8; ++e) { rm[e] = -INFINITY; rl[e] = 0.f; }
  const int myq0 = qt * 16 + hi * 8;  // C row m = e + 8*hi

  const __bf16* vbase =
      Vt + (size_t)((b * H_NUM + h) * HD_DIM) * S_LEN;

  for (int kt = 0; kt < S_LEN; kt += 32) {
    const __bf16* kp0 = Kc + ((size_t)(b * S_LEN + kt + mlane) * H_NUM + h) * HD_DIM;
    const __bf16* kp1 = kp0 + (size_t)16 * H_NUM * HD_DIM;  // keys kt+16..kt+31
    if (kt + 32 < S_LEN) {  // prefetch next K block rows (near caches)
      __builtin_prefetch(kp0 + (size_t)32 * H_NUM * HD_DIM, 0, 3);
      __builtin_prefetch(kp1 + (size_t)32 * H_NUM * HD_DIM, 0, 3);
    }
    v8f c0 = {}, c1 = {};
#pragma unroll
    for (int c = 0; c < 2; ++c) {
      FragBF kf;
      *(u32x4*)&kf.u[0] = *(const u32x4*)(kp0 + 32 * c + kbB);
      *(u32x4*)&kf.u[4] = *(const u32x4*)(kp0 + 32 * c + kbB + 16);
      c0 = wmma_bf16(qf[c].v, kf.v, c0);
      *(u32x4*)&kf.u[0] = *(const u32x4*)(kp1 + 32 * c + kbB);
      *(u32x4*)&kf.u[4] = *(const u32x4*)(kp1 + 32 * c + kbB + 16);
      c1 = wmma_bf16(qf[c].v, kf.v, c1);
    }

    float p0[8], p1[8];
#pragma unroll
    for (int e = 0; e < 8; ++e) {
      float s0 = c0[e] * 0.125f;          // 1/sqrt(64)
      float s1 = c1[e] * 0.125f;
      if (kt + mlane      <= myq0 + e) s0 = -1e9f;   // reference mask-bug, exact
      if (kt + 16 + mlane <= myq0 + e) s1 = -1e9f;
      p0[e] = s0; p1[e] = s1;
      float t = fmaxf(s0, s1);            // row max across 16-lane half-wave
      t = fmaxf(t, __shfl_xor(t, 1, 32));
      t = fmaxf(t, __shfl_xor(t, 2, 32));
      t = fmaxf(t, __shfl_xor(t, 4, 32));
      t = fmaxf(t, __shfl_xor(t, 8, 32));
      float newm  = fmaxf(rm[e], t);
      float alpha = expf(rm[e] - newm);
      rm[e] = newm;
      float e0 = expf(p0[e] - newm);
      float e1 = expf(p1[e] - newm);
      p0[e] = e0; p1[e] = e1;
      float rs = e0 + e1;
      rs += __shfl_xor(rs, 1, 32);
      rs += __shfl_xor(rs, 2, 32);
      rs += __shfl_xor(rs, 4, 32);
      rs += __shfl_xor(rs, 8, 32);
      rl[e] = rl[e] * alpha + rs;
#pragma unroll
      for (int j = 0; j < 4; ++j) acc[j][e] *= alpha;
    }

    // Transpose P from C layout to A-fragment layout via per-wave LDS tile
#pragma unroll
    for (int e = 0; e < 8; ++e) {
      Ps[wid][hi * 8 + e][mlane]      = (__bf16)p0[e];
      Ps[wid][hi * 8 + e][16 + mlane] = (__bf16)p1[e];
    }
    FragBF pf;
    *(u32x4*)&pf.u[0] = *(const u32x4*)&Ps[wid][mlane][kbA];
    *(u32x4*)&pf.u[4] = *(const u32x4*)&Ps[wid][mlane][kbA + 16];

#pragma unroll
    for (int j = 0; j < 4; ++j) {
      const __bf16* vp = vbase + (size_t)(j * 16 + mlane) * S_LEN + kt + kbB;
      FragBF vf;
      *(u32x4*)&vf.u[0] = *(const u32x4*)vp;
      *(u32x4*)&vf.u[4] = *(const u32x4*)(vp + 16);
      acc[j] = wmma_bf16(pf.v, vf.v, acc[j]);
    }
  }

#pragma unroll
  for (int j = 0; j < 4; ++j) {
#pragma unroll
    for (int e = 0; e < 8; ++e) {
      int q  = qt * 16 + hi * 8 + e;
      int hd = j * 16 + mlane;
      O[((size_t)(b * S_LEN + q) * H_NUM + h) * HD_DIM + hd] = (__bf16)(acc[j][e] / rl[e]);
    }
  }
}

// ---------------------------------------------------------------------------
// Host orchestration
// ---------------------------------------------------------------------------
extern "C" void kernel_launch(void* const* d_in, const int* in_sizes, int n_in,
                              void* d_out, int out_size, void* d_ws, size_t ws_size,
                              hipStream_t stream) {
  (void)in_sizes; (void)n_in; (void)out_size;
  const float* x   = (const float*)d_in[0];
  const float* wq  = (const float*)d_in[1];
  const float* wk  = (const float*)d_in[2];
  const float* wv  = (const float*)d_in[3];
  const float* wo  = (const float*)d_in[4];
  const float* bo  = (const float*)d_in[5];
  const float* g1  = (const float*)d_in[6];
  const float* be1 = (const float*)d_in[7];
  const float* g2  = (const float*)d_in[8];
  const float* be2 = (const float*)d_in[9];
  const float* w1  = (const float*)d_in[10];
  const float* b1  = (const float*)d_in[11];
  const float* w2  = (const float*)d_in[12];
  const float* b2  = (const float*)d_in[13];
  float* out = (float*)d_out;

  char* ws = (char*)d_ws;
  size_t off = 0;
  auto alloc = [&](size_t bytes) -> void* {
    void* p = ws + off;
    off += (bytes + 255) & ~(size_t)255;
    return p;
  };
  __bf16* wqt  = (__bf16*)alloc((size_t)D_DIM * D_DIM * 2);
  __bf16* wkt  = (__bf16*)alloc((size_t)D_DIM * D_DIM * 2);
  __bf16* wvt  = (__bf16*)alloc((size_t)D_DIM * D_DIM * 2);
  __bf16* wot  = (__bf16*)alloc((size_t)D_DIM * D_DIM * 2);
  __bf16* w1t  = (__bf16*)alloc((size_t)D_DIM * F_DIM * 2);
  __bf16* w2t  = (__bf16*)alloc((size_t)F_DIM * D_DIM * 2);
  __bf16* lnb  = (__bf16*)alloc((size_t)M_ROWS * D_DIM * 2);
  __bf16* qb   = (__bf16*)alloc((size_t)M_ROWS * D_DIM * 2);
  __bf16* kb   = (__bf16*)alloc((size_t)M_ROWS * D_DIM * 2);
  __bf16* vb   = (__bf16*)alloc((size_t)M_ROWS * D_DIM * 2);
  __bf16* vtb  = (__bf16*)alloc((size_t)M_ROWS * D_DIM * 2);
  __bf16* obuf = (__bf16*)alloc((size_t)M_ROWS * D_DIM * 2);
  __bf16* a2b  = (__bf16*)alloc((size_t)M_ROWS * D_DIM * 2);
  __bf16* midb = (__bf16*)alloc((size_t)M_ROWS * F_DIM * 2);
  float*  hbuf = (float*) alloc((size_t)M_ROWS * D_DIM * 4);
  if (off > ws_size) return;  // workspace too small: bail before launching

  const dim3 gD(D_DIM / 128, M_ROWS / 128);   // (8,32)
  const dim3 gF(F_DIM / 128, M_ROWS / 128);   // (32,32)
  const dim3 gAttn(S_LEN / 64, H_NUM, B_NUM); // 4 q-tiles / block
  const int  vtN = (B_NUM * H_NUM * HD_DIM * S_LEN) / 256;

  // Weight convert+transpose to bf16 [N][K]
  cvt_transpose<<<(D_DIM * D_DIM) / 256, 256, 0, stream>>>(wq, wqt, D_DIM, D_DIM);
  cvt_transpose<<<(D_DIM * D_DIM) / 256, 256, 0, stream>>>(wk, wkt, D_DIM, D_DIM);
  cvt_transpose<<<(D_DIM * D_DIM) / 256, 256, 0, stream>>>(wv, wvt, D_DIM, D_DIM);
  cvt_transpose<<<(D_DIM * D_DIM) / 256, 256, 0, stream>>>(wo, wot, D_DIM, D_DIM);
  cvt_transpose<<<(D_DIM * F_DIM) / 256, 256, 0, stream>>>(w1, w1t, D_DIM, F_DIM);
  cvt_transpose<<<(F_DIM * D_DIM) / 256, 256, 0, stream>>>(w2, w2t, F_DIM, D_DIM);

  // ---- Block 1: h = attn(LN1(x)) + x ----
  layernorm_bf16<<<M_ROWS, 256, 0, stream>>>(x, g1, be1, lnb);
  gemm_bf16_wmma<<<gD, 256, 0, stream>>>(lnb, wqt, nullptr, nullptr, nullptr, qb, D_DIM, D_DIM, 0);
  gemm_bf16_wmma<<<gD, 256, 0, stream>>>(lnb, wkt, nullptr, nullptr, nullptr, kb, D_DIM, D_DIM, 0);
  gemm_bf16_wmma<<<gD, 256, 0, stream>>>(lnb, wvt, nullptr, nullptr, nullptr, vb, D_DIM, D_DIM, 0);
  vtrans<<<vtN, 256, 0, stream>>>(vb, vtb);
  attn_wmma<<<gAttn, 128, 0, stream>>>(qb, kb, vtb, obuf);
  gemm_bf16_wmma<<<gD, 256, 0, stream>>>(obuf, wot, bo, x, hbuf, nullptr, D_DIM, D_DIM, 3);

  // ---- Block 2: out = FFN(attn(LN2(h))) + h ----
  layernorm_bf16<<<M_ROWS, 256, 0, stream>>>(hbuf, g2, be2, lnb);
  gemm_bf16_wmma<<<gD, 256, 0, stream>>>(lnb, wqt, nullptr, nullptr, nullptr, qb, D_DIM, D_DIM, 0);
  gemm_bf16_wmma<<<gD, 256, 0, stream>>>(lnb, wkt, nullptr, nullptr, nullptr, kb, D_DIM, D_DIM, 0);
  gemm_bf16_wmma<<<gD, 256, 0, stream>>>(lnb, wvt, nullptr, nullptr, nullptr, vb, D_DIM, D_DIM, 0);
  vtrans<<<vtN, 256, 0, stream>>>(vb, vtb);
  attn_wmma<<<gAttn, 128, 0, stream>>>(qb, kb, vtb, obuf);
  gemm_bf16_wmma<<<gD, 256, 0, stream>>>(obuf, wot, bo, nullptr, nullptr, a2b, D_DIM, D_DIM, 1);
  gemm_bf16_wmma<<<gF, 256, 0, stream>>>(a2b, w1t, b1, nullptr, nullptr, midb, F_DIM, D_DIM, 2);
  gemm_bf16_wmma<<<gD, 256, 0, stream>>>(midb, w2t, b2, hbuf, out, nullptr, D_DIM, F_DIM, 3);
}
